// StandardMultiHeadAttention_2353642078532
// MI455X (gfx1250) — matrix operand order repaired and checked by reading, not compiled, not run on hardware
//
#include <hip/hip_runtime.h>
#include <hip/hip_bf16.h>

typedef __attribute__((ext_vector_type(16))) _Float16 v16h;
typedef __attribute__((ext_vector_type(8)))  _Float16 v8h;
typedef __attribute__((ext_vector_type(8)))  float    v8f;

#define WMMA_F16(a, b, c) \
  __builtin_amdgcn_wmma_f32_16x16x32_f16(false, (a), false, (b), (short)0, (c), false, false)

// ---------------- Tensor Data Mover (CDNA5 TDM) ----------------
#if defined(__AMDGCN__) && __has_builtin(__builtin_amdgcn_tensor_load_to_lds)
#define HAVE_TDM 1
typedef __attribute__((ext_vector_type(4))) unsigned int v4u;
typedef __attribute__((ext_vector_type(8))) int v8i;
typedef __attribute__((ext_vector_type(4))) int v4i;

// 2D tile load: LDS[lds_addr + (y*tile_d0 + x)*2] = gsrc[y*stride_d0 + x]  (f16)
// D# layout per CDNA5 ISA ch.8: group0 = {count/flags, lds_addr, global_addr, type},
// group1 = {mask/data_size, tensor dims, tile dims, dim0 stride}.
// This toolchain exposes the 6-arg builtin (clang-23 / therock form).
__device__ __forceinline__ void tdm_load_tile_f16(const _Float16* gsrc, unsigned lds_addr,
                                                  unsigned tensor_d0, unsigned tensor_d1,
                                                  unsigned tile_d0, unsigned tile_d1,
                                                  unsigned stride_d0) {
  unsigned long long ga = (unsigned long long)(uintptr_t)gsrc;
  v4u g0;
  g0[0] = 1u;                                               // count=1, user mode
  g0[1] = lds_addr;                                         // LDS byte address
  g0[2] = (unsigned)(ga & 0xffffffffu);                     // global_addr[31:0]
  g0[3] = (unsigned)((ga >> 32) & 0x01ffffffu) | (2u << 30);// global_addr[56:32] | type=2
  v8i g1;
  g1[0] = (int)(1u << 16);                                  // wg_mask=0, data_size=1 (2B)
  g1[1] = (int)((tensor_d0 & 0xffffu) << 16);               // tensor_dim0[15:0]
  g1[2] = (int)((tensor_d0 >> 16) | ((tensor_d1 & 0xffffu) << 16));
  g1[3] = (int)((tensor_d1 >> 16) | (tile_d0 << 16));       // tile_dim0
  g1[4] = (int)tile_d1;                                     // tile_dim1, tile_dim2=0
  g1[5] = (int)stride_d0;                                   // tensor_dim0_stride[31:0]
  g1[6] = 0;                                                // stride0 hi, stride1 lo
  g1[7] = 0;                                                // stride1 hi
  v4i g2 = {0, 0, 0, 0};                                    // group2 (unused for 2D tile)
  v4i g3 = {0, 0, 0, 0};                                    // group3 (unused for 2D tile)
  v8i g4 = {0, 0, 0, 0, 0, 0, 0, 0};                        // extra group (zero)
  __builtin_amdgcn_tensor_load_to_lds(g0, g1, g2, g3, g4, 0);
}
#endif

// ---- WMMA fragment loaders (per CDNA5 ISA 7.12.2 16-bit layouts) ----
// A 16x32 (MxK): rowptr already points at [m=lane&15][k_chunk_base]; element (m,k).
__device__ __forceinline__ v16h frag_a_row(const _Float16* rowptr, int lane) {
  const int k0 = (lane < 16) ? 0 : 8;
  union { v16h v; v8h h[2]; } u;
  u.h[0] = *(const v8h*)(rowptr + k0);
  u.h[1] = *(const v8h*)(rowptr + k0 + 16);
  return u.v;
}
// B 32x16 (KxN): colptr points at column n=lane&15 stored K-contiguous.
__device__ __forceinline__ v16h frag_b_col(const _Float16* colptr, int lane) {
  const int k0 = (lane < 16) ? 0 : 8;
  union { v16h v; v8h h[2]; } u;
  u.h[0] = *(const v8h*)(colptr + k0);
  u.h[1] = *(const v8h*)(colptr + k0 + 16);
  return u.v;
}

// ---------------- f32 -> f16 convert (optionally scaled) ----------------
__global__ void k_cvt(const float* __restrict__ in, _Float16* __restrict__ out,
                      float scale, int n) {
  int i = blockIdx.x * blockDim.x + threadIdx.x;
  if (i < n) out[i] = (_Float16)(in[i] * scale);
}

// ------------- build transposed conv band matrices Art[r][jout][jin] -------------
__global__ void k_conv_mat(const float* __restrict__ c1k, const float* __restrict__ c3k,
                           const float* __restrict__ c5k, _Float16* __restrict__ Art) {
  int r    = blockIdx.x >> 8;    // 0..4  (row offset r-2 in [-2,2])
  int jout = blockIdx.x & 255;
  int jin  = threadIdx.x;
  int w = jin - jout + 127;
  float v = 0.0f;
  if (w >= 0 && w < 256) {
    v = c5k[r * 256 + w];
    if (r >= 1 && r <= 3) v += c3k[(r - 1) * 256 + w];
    if (r == 2) v += c1k[w];
  }
  Art[(size_t)blockIdx.x * 256 + jin] = (_Float16)(v * (1.0f / 3.0f));
}

// ---------------- pairwise features + tiny MLP (head-independent) ----------------
__global__ void k_mpair(const float* __restrict__ particles,
                        const float* __restrict__ mw1, const float* __restrict__ mb1,
                        const float* __restrict__ mw2, const float* __restrict__ mb2,
                        const float* __restrict__ mw3, const float* __restrict__ mb3,
                        _Float16* __restrict__ mp) {
  int idx = blockIdx.x * 256 + threadIdx.x;     // b*65536 + i*256 + j
  int j = idx & 255;
  int i = (idx >> 8) & 255;
  int b = idx >> 16;
  const float* Pi = particles + ((size_t)b * 256 + i) * 3;
  const float* Pj = particles + ((size_t)b * 256 + j) * 3;
  float pti = Pi[0], etai = Pi[1], phii = Pi[2];
  float ptj = Pj[0], etaj = Pj[1], phij = Pj[2];
  float d_eta = etai - etaj;
  float dp = phii - phij;
  float sdp = sinf(dp), cdp = cosf(dp);
  float d_phi = atan2f(sdp, cdp);
  float delta = sqrtf(d_eta * d_eta + d_phi * d_phi);
  float minpt = fminf(pti, ptj);
  float kT = minpt * delta;
  float z = minpt / (pti + ptj + 1e-8f);
  float m2 = fmaxf(pti * pti + ptj * ptj - 2.0f * pti * ptj * cdp, 1e-8f);
  float f[4];
  f[0] = logf(fmaxf(delta, 1e-8f));
  f[1] = logf(fmaxf(kT, 1e-8f));
  f[2] = logf(fmaxf(z, 1e-8f));
  f[3] = logf(m2);
  float h1[8], h2[8];
#pragma unroll
  for (int o = 0; o < 8; ++o) {
    float s = mb1[o];
#pragma unroll
    for (int q = 0; q < 4; ++q) s += f[q] * mw1[q * 8 + o];
    h1[o] = fmaxf(s, 0.0f);
  }
#pragma unroll
  for (int o = 0; o < 8; ++o) {
    float s = mb2[o];
#pragma unroll
    for (int q = 0; q < 8; ++q) s += h1[q] * mw2[q * 8 + o];
    h2[o] = fmaxf(s, 0.0f);
  }
  float s = mb3[0];
#pragma unroll
  for (int q = 0; q < 8; ++q) s += h2[q] * mw3[q];
  s = fminf(fmaxf(s, -10.0f), 10.0f);
  mp[idx] = (_Float16)s;
}

// ---------------- generic 128x128-tile WMMA GEMM, double-buffered ----------------
// A tile (128x32, layout-preserving) staged by the Tensor Data Mover; B tile staged
// manually (needs [n][k] transpose for the B fragment).  One barrier per K-step.
// mode 0: f16 store out[row*N+col]
// mode 1: f16 store V transposed per head: out[(b*8+h)*64+d][s]
// mode 2: f32 store + bias (final dense)
__global__ __launch_bounds__(256) void k_gemm(const _Float16* __restrict__ A,
                                              const _Float16* __restrict__ Bw,
                                              void* __restrict__ outp,
                                              const float* __restrict__ bias,
                                              int M, int N, int K, int mode) {
  __shared__ _Float16 At[2][128 * 32];   // [buf][m][k]
  __shared__ _Float16 BT[2][128 * 32];   // [buf][n][k]
  const int tid = threadIdx.x;
  const int lane = tid & 31;
  const int wid = tid >> 5;              // 8 waves: 2 (m) x 4 (n)
  const int wm = wid & 1, wn = wid >> 1;
  const int nbn = N >> 7;
  const int row0 = (blockIdx.x / nbn) << 7;
  const int col0 = (blockIdx.x % nbn) << 7;

  v8f acc[4][2] = {};
  const int ar = tid >> 1, as = (tid & 1) << 4;      // A fallback loader coords
  const int bk = tid & 31, bn0 = (tid >> 5) << 4;    // B tile loader coords

  const _Float16* Abase = A + (size_t)row0 * K;

  // ---- prologue: stage K-tile 0 into buffer 0 ----
#ifdef HAVE_TDM
  if (wid == 0)
    tdm_load_tile_f16(Abase, (unsigned)(uintptr_t)&At[0][0],
                      (unsigned)K, (unsigned)M, 32u, 128u, (unsigned)K);
#else
  {
    const v8h* asrc = (const v8h*)(Abase + (size_t)ar * K + as);
    *(v8h*)&At[0][ar * 32 + as] = asrc[0];
    *(v8h*)&At[0][ar * 32 + as + 8] = asrc[1];
  }
#endif
  {
    const _Float16* bsrc = Bw + (size_t)bk * N + col0 + bn0;
    v8h b0 = *(const v8h*)bsrc;
    v8h b1 = *(const v8h*)(bsrc + 8);
#pragma unroll
    for (int e = 0; e < 8; ++e) BT[0][(bn0 + e) * 32 + bk] = b0[e];
#pragma unroll
    for (int e = 0; e < 8; ++e) BT[0][(bn0 + 8 + e) * 32 + bk] = b1[e];
  }
#ifdef HAVE_TDM
  if (wid == 0) __builtin_amdgcn_s_wait_tensorcnt(0);
#endif
  __syncthreads();

  int cur = 0;
  for (int kb = 0; kb < K; kb += 32) {
    const int nxt = cur ^ 1;
    const bool have_next = (kb + 32 < K);

    // kick off next tile's movement first: TDM for A, global->regs for B
    v8h nb0, nb1;
    if (have_next) {
#ifdef HAVE_TDM
      if (wid == 0)
        tdm_load_tile_f16(Abase + kb + 32, (unsigned)(uintptr_t)&At[nxt][0],
                          (unsigned)K, (unsigned)M, 32u, 128u, (unsigned)K);
#else
      const v8h* asrc = (const v8h*)(Abase + (size_t)ar * K + kb + 32 + as);
      v8h na0 = asrc[0], na1 = asrc[1];
      *(v8h*)&At[nxt][ar * 32 + as] = na0;
      *(v8h*)&At[nxt][ar * 32 + as + 8] = na1;
#endif
      const _Float16* bsrc = Bw + (size_t)(kb + 32 + bk) * N + col0 + bn0;
      nb0 = *(const v8h*)bsrc;
      nb1 = *(const v8h*)(bsrc + 8);
      if (kb + 64 < K)  // global_prefetch_b8 two tiles ahead (B operand)
        __builtin_prefetch(Bw + (size_t)(kb + 64 + bk) * N + col0 + bn0, 0, 1);
    }

    // compute on current buffers
    v16h bf0 = frag_b_col(&BT[cur][(wn * 32 + (lane & 15)) * 32], lane);
    v16h bf1 = frag_b_col(&BT[cur][(wn * 32 + 16 + (lane & 15)) * 32], lane);
#pragma unroll
    for (int mt = 0; mt < 4; ++mt) {
      v16h af = frag_a_row(&At[cur][(wm * 64 + mt * 16 + (lane & 15)) * 32], lane);
      acc[mt][0] = WMMA_F16(af, bf0, acc[mt][0]);
      acc[mt][1] = WMMA_F16(af, bf1, acc[mt][1]);
    }

    if (have_next) {
#pragma unroll
      for (int e = 0; e < 8; ++e) BT[nxt][(bn0 + e) * 32 + bk] = nb0[e];
#pragma unroll
      for (int e = 0; e < 8; ++e) BT[nxt][(bn0 + 8 + e) * 32 + bk] = nb1[e];
#ifdef HAVE_TDM
      if (wid == 0) __builtin_amdgcn_s_wait_tensorcnt(0);
#endif
    }
    __syncthreads();
    cur = nxt;
  }

  const int mb_ = (lane < 16) ? 0 : 8;
  const int cn = lane & 15;
  if (mode == 2) {
    float* out = (float*)outp;
#pragma unroll
    for (int mt = 0; mt < 4; ++mt)
#pragma unroll
      for (int nt = 0; nt < 2; ++nt) {
        int col = col0 + wn * 32 + nt * 16 + cn;
        float bv = bias[col];
#pragma unroll
        for (int g = 0; g < 8; ++g) {
          int row = row0 + wm * 64 + mt * 16 + mb_ + g;
          out[(size_t)row * N + col] = acc[mt][nt][g] + bv;
        }
      }
  } else if (mode == 1) {
    _Float16* out = (_Float16*)outp;
#pragma unroll
    for (int mt = 0; mt < 4; ++mt)
#pragma unroll
      for (int nt = 0; nt < 2; ++nt) {
        int col = col0 + wn * 32 + nt * 16 + cn;
        int hh = col >> 6, d = col & 63;
#pragma unroll
        for (int g = 0; g < 8; ++g) {
          int row = row0 + wm * 64 + mt * 16 + mb_ + g;
          int bb = row >> 8, s = row & 255;
          out[(((size_t)(bb * 8 + hh) * 64 + d) << 8) | s] = (_Float16)acc[mt][nt][g];
        }
      }
  } else {
    _Float16* out = (_Float16*)outp;
#pragma unroll
    for (int mt = 0; mt < 4; ++mt)
#pragma unroll
      for (int nt = 0; nt < 2; ++nt) {
        int col = col0 + wn * 32 + nt * 16 + cn;
#pragma unroll
        for (int g = 0; g < 8; ++g) {
          int row = row0 + wm * 64 + mt * 16 + mb_ + g;
          out[(size_t)row * N + col] = (_Float16)acc[mt][nt][g];
        }
      }
  }
}

// ---------------- fused attention: scores -> conv -> +m -> softmax -> P@V ----------------
// grid: (B*H) * (S/32) blocks of 128 threads (4 waves)
__global__ __launch_bounds__(128) void k_attn(const _Float16* __restrict__ Qh,
                                              const _Float16* __restrict__ Kh,
                                              const _Float16* __restrict__ Vt,
                                              const _Float16* __restrict__ Art,
                                              const _Float16* __restrict__ mp,
                                              const float* __restrict__ c1b,
                                              const float* __restrict__ c3b,
                                              const float* __restrict__ c5b,
                                              _Float16* __restrict__ attn_out) {
  __shared__ _Float16 Sraw[64 * 256];   // raw scores rows [i0-16, i0+48), f16
  __shared__ _Float16 Tbuf[32 * 256];   // conv+bias scores, then softmax probs
  const int qblk = blockIdx.x & 7;
  const int bh   = blockIdx.x >> 3;
  const int b = bh >> 3, h = bh & 7;
  const int i0 = qblk * 32;
  const int tid = threadIdx.x;
  const int wv = tid >> 5;
  const int lane = tid & 31;
  const int m15 = lane & 15;

  const _Float16* Qbase = Qh + (size_t)b * 256 * 512 + h * 64;
  const _Float16* Kbase = Kh + (size_t)b * 256 * 512 + h * 64;

  // ---- Phase A: raw scores S = (Q/sqrt(dh)) K^T for 64 rows (16-row halo each side) ----
  {
    int gq = i0 - 16 + wv * 16 + m15;
    int gqc = gq < 0 ? 0 : (gq > 255 ? 255 : gq);
    const _Float16* arow = Qbase + (size_t)gqc * 512;
    for (int nt = 0; nt < 16; ++nt) {
      const _Float16* bcol = Kbase + (size_t)(nt * 16 + m15) * 512;
      v8f acc = {};
#pragma unroll
      for (int ks = 0; ks < 2; ++ks) {
        v16h af = frag_a_row(arow + ks * 32, lane);
        v16h bf = frag_b_col(bcol + ks * 32, lane);
        acc = WMMA_F16(af, bf, acc);
      }
      int mbase = (lane < 16) ? 0 : 8;
      int col = nt * 16 + m15;
#pragma unroll
      for (int g = 0; g < 8; ++g)
        Sraw[(wv * 16 + mbase + g) * 256 + col] = (_Float16)acc[g];
    }
  }
  __syncthreads();
  // zero padding rows (conv SAME pads with zeros beyond the 256-row tensor)
  if (qblk == 0 || qblk == 7) {
    for (int idx = tid; idx < 64 * 256; idx += 128) {
      int gq = i0 - 16 + (idx >> 8);
      if (gq < 0 || gq >= 256) Sraw[idx] = (_Float16)0.0f;
    }
    __syncthreads();
  }

  // ---- Phase B: T[i,:] = sum_{r=-2..2} Sraw[i+r,:] @ A_r  (+ bias + pairwise m) ----
  {
    float cb = (c1b[0] + c3b[0] + c5b[0]) * (1.0f / 3.0f);
    v8f acc[2][4] = {};   // 2 m-tiles (32 rows) x 4 n-tiles (wave's 64 cols)
    for (int r = 0; r < 5; ++r) {
      const _Float16* ArtR = Art + (size_t)r * 256 * 256;
#pragma unroll
      for (int ks = 0; ks < 8; ++ks) {
        int kk = ks * 32;
        // local T row block starts at Sraw row 16; shift by r-2
        v16h a0 = frag_a_row(&Sraw[(14 + r + m15) * 256 + kk], lane);
        v16h a1 = frag_a_row(&Sraw[(14 + r + 16 + m15) * 256 + kk], lane);
#pragma unroll
        for (int nt = 0; nt < 4; ++nt) {
          int col = (wv * 4 + nt) * 16 + m15;
          v16h bf = frag_b_col(ArtR + (size_t)col * 256 + kk, lane);
          acc[0][nt] = WMMA_F16(a0, bf, acc[0][nt]);
          acc[1][nt] = WMMA_F16(a1, bf, acc[1][nt]);
        }
      }
    }
    const _Float16* mrow = mp + ((size_t)b * 256 + i0) * 256;
    int mbase = (lane < 16) ? 0 : 8;
#pragma unroll
    for (int mt = 0; mt < 2; ++mt)
#pragma unroll
      for (int nt = 0; nt < 4; ++nt) {
        int col = (wv * 4 + nt) * 16 + m15;
#pragma unroll
        for (int g = 0; g < 8; ++g) {
          int row = mt * 16 + mbase + g;   // local 0..31, global q = i0+row
          float v = acc[mt][nt][g] + cb + (float)mrow[(size_t)row * 256 + col];
          Tbuf[row * 256 + col] = (_Float16)v;
        }
      }
  }
  __syncthreads();

  // ---- Phase C: softmax over k (256 cols); probs overwrite Tbuf in f16 ----
  if (tid < 32) {
    float mx = -1e30f;
    for (int jx = 0; jx < 256; ++jx) mx = fmaxf(mx, (float)Tbuf[tid * 256 + jx]);
    float sum = 0.0f;
    for (int jx = 0; jx < 256; ++jx) sum += __expf((float)Tbuf[tid * 256 + jx] - mx);
    float inv = 1.0f / sum;
    for (int jx = 0; jx < 256; ++jx)
      Tbuf[tid * 256 + jx] = (_Float16)(__expf((float)Tbuf[tid * 256 + jx] - mx) * inv);
  }
  __syncthreads();

  // ---- Phase D: O = P(32x256) @ V(256x64); wave wv owns 16 output cols ----
  {
    v8f o0 = {}, o1 = {};
    const _Float16* vcol = Vt + ((size_t)bh * 64 + wv * 16 + m15) * 256;
#pragma unroll
    for (int ks = 0; ks < 8; ++ks) {
      int kk = ks * 32;
      v16h bf = frag_b_col(vcol + kk, lane);
      v16h a0 = frag_a_row(&Tbuf[m15 * 256 + kk], lane);
      v16h a1 = frag_a_row(&Tbuf[(16 + m15) * 256 + kk], lane);
      o0 = WMMA_F16(a0, bf, o0);
      o1 = WMMA_F16(a1, bf, o1);
    }
    int mbase = (lane < 16) ? 0 : 8;
    int col = h * 64 + wv * 16 + m15;
#pragma unroll
    for (int g = 0; g < 8; ++g) {
      int s0 = i0 + mbase + g;
      int s1 = i0 + 16 + mbase + g;
      attn_out[((size_t)(b * 256 + s0)) * 512 + col] = (_Float16)o0[g];
      attn_out[((size_t)(b * 256 + s1)) * 512 + col] = (_Float16)o1[g];
    }
  }
}

extern "C" void kernel_launch(void* const* d_in, const int* in_sizes, int n_in,
                              void* d_out, int out_size, void* d_ws, size_t ws_size,
                              hipStream_t stream) {
  const float* x         = (const float*)d_in[0];
  const float* particles = (const float*)d_in[1];
  const float* wq        = (const float*)d_in[2];
  const float* wk        = (const float*)d_in[3];
  const float* wv        = (const float*)d_in[4];
  const float* dense_w   = (const float*)d_in[5];
  const float* dense_b   = (const float*)d_in[6];
  const float* c1k       = (const float*)d_in[7];
  const float* c1b       = (const float*)d_in[8];
  const float* c3k       = (const float*)d_in[9];
  const float* c3b       = (const float*)d_in[10];
  const float* c5k       = (const float*)d_in[11];
  const float* c5b       = (const float*)d_in[12];
  const float* mw1       = (const float*)d_in[13];
  const float* mb1       = (const float*)d_in[14];
  const float* mw2       = (const float*)d_in[15];
  const float* mb2       = (const float*)d_in[16];
  const float* mw3       = (const float*)d_in[17];
  const float* mb3       = (const float*)d_in[18];

  const size_t NX = (size_t)16384 * 512;   // B*S x D
  const size_t NW = (size_t)512 * 512;
  _Float16* wsp = (_Float16*)d_ws;
  _Float16* xh  = wsp;
  _Float16* wqh = xh + NX;
  _Float16* wkh = wqh + NW;
  _Float16* wvh = wkh + NW;
  _Float16* dwh = wvh + NW;
  _Float16* Qh  = dwh + NW;
  _Float16* Kh  = Qh + NX;
  _Float16* Vt  = Kh + NX;                       // [b][h][d][s]
  _Float16* Art = Vt + NX;                       // 5 x 256 x 256
  _Float16* mp  = Art + (size_t)5 * 256 * 256;   // 64 x 256 x 256
  _Float16* ah  = mp + (size_t)64 * 256 * 256;   // attn concat f16

  k_cvt<<<(int)((NX + 255) / 256), 256, 0, stream>>>(x, xh, 1.0f, (int)NX);
  k_cvt<<<(int)((NW + 255) / 256), 256, 0, stream>>>(wq, wqh, 0.125f, (int)NW); // fold 1/sqrt(64)
  k_cvt<<<(int)((NW + 255) / 256), 256, 0, stream>>>(wk, wkh, 1.0f, (int)NW);
  k_cvt<<<(int)((NW + 255) / 256), 256, 0, stream>>>(wv, wvh, 1.0f, (int)NW);
  k_cvt<<<(int)((NW + 255) / 256), 256, 0, stream>>>(dense_w, dwh, 1.0f, (int)NW);

  k_conv_mat<<<5 * 256, 256, 0, stream>>>(c1k, c3k, c5k, Art);
  k_mpair<<<64 * 256, 256, 0, stream>>>(particles, mw1, mb1, mw2, mb2, mw3, mb3, mp);

  k_gemm<<<512, 256, 0, stream>>>(xh, wqh, Qh, nullptr, 16384, 512, 512, 0);
  k_gemm<<<512, 256, 0, stream>>>(xh, wkh, Kh, nullptr, 16384, 512, 512, 0);
  k_gemm<<<512, 256, 0, stream>>>(xh, wvh, Vt, nullptr, 16384, 512, 512, 1);

  k_attn<<<4096, 128, 0, stream>>>(Qh, Kh, Vt, Art, mp, c1b, c3b, c5b, ah);

  k_gemm<<<512, 256, 0, stream>>>(ah, dwh, d_out, dense_b, 16384, 512, 512, 2);
}
